// TransformerEncoderLayer_2216203125583
// MI455X (gfx1250) — compile-verified, matrix operand order checked by reading, not run on hardware
//
#include <hip/hip_runtime.h>
#include <cstddef>

// ---------------- problem constants ----------------
constexpr int B_ = 8, S_ = 1024, D_ = 768, H_ = 12, DK_ = 64, DFF_ = 3072;
constexpr size_t NTOK = (size_t)B_ * S_;           // 8192 token rows

typedef _Float16 half_t;
typedef __attribute__((ext_vector_type(16))) _Float16 v16h;
typedef __attribute__((ext_vector_type(8)))  _Float16 v8h;
typedef __attribute__((ext_vector_type(8)))  float    v8f;
typedef __attribute__((ext_vector_type(4)))  int      v4i;

// Async global->LDS copy (gfx1250 GLOBAL_LOAD_ASYNC_TO_LDS_B128), if the
// toolchain exposes the builtin; otherwise fall back to load+ds_store.
#if defined(__has_builtin)
#if __has_builtin(__builtin_amdgcn_global_load_async_to_lds_b128)
#define USE_ASYNC_LDS 1
#endif
#endif

#ifdef USE_ASYNC_LDS
typedef __attribute__((address_space(1))) v4i gv4i;   // global 128-bit chunk
typedef __attribute__((address_space(3))) v4i lv4i;   // LDS 128-bit chunk
__device__ __forceinline__ void async_cp16(const half_t* g, half_t* l) {
    __builtin_amdgcn_global_load_async_to_lds_b128((gv4i*)g, (lv4i*)l, 0, 0);
}
#if __has_builtin(__builtin_amdgcn_s_wait_asynccnt)
#define WAIT_ASYNC0() __builtin_amdgcn_s_wait_asynccnt(0)
#else
#define WAIT_ASYNC0() asm volatile("s_wait_asynccnt 0" ::: "memory")
#endif
#endif

// ---------------- WMMA helpers ----------------
__device__ __forceinline__ v8f wmma_f16(v16h a, v16h b, v8f c) {
    // D(16x16,f32) = A(16x32,f16) * B(32x16,f16) + C
    return __builtin_amdgcn_wmma_f32_16x16x32_f16(
        /*neg_a=*/false, a, /*neg_b=*/false, b,
        /*c_mod=*/(short)0, c, /*reuse_a=*/false, /*reuse_b=*/false);
}

// Load an A- or B-fragment for v_wmma_f32_16x16x32_f16.
// Per ISA layout: lane (li = lane&15) owns row/col li; K values are two
// contiguous 8-half (16B) chunks at kbase + hi*8 and kbase + 16 + hi*8.
__device__ __forceinline__ v16h load_frag(const half_t* base, int ld, int kbase) {
    const int lane = threadIdx.x & 31;
    const int li = lane & 15;
    const int hi = lane >> 4;
    const half_t* p = base + (size_t)li * ld + kbase + hi * 8;
    v8h c0 = *(const v8h*)(p);
    v8h c1 = *(const v8h*)(p + 16);
    return __builtin_shufflevector(c0, c1, 0,1,2,3,4,5,6,7,8,9,10,11,12,13,14,15);
}

// 16-lane-half reductions (wave32: rows 0..7 live in lanes 0-15, 8..15 in 16-31)
__device__ __forceinline__ float hmax16(float v) {
    v = fmaxf(v, __shfl_xor(v, 1, 32));
    v = fmaxf(v, __shfl_xor(v, 2, 32));
    v = fmaxf(v, __shfl_xor(v, 4, 32));
    v = fmaxf(v, __shfl_xor(v, 8, 32));
    return v;
}
__device__ __forceinline__ float hsum16(float v) {
    v += __shfl_xor(v, 1, 32);
    v += __shfl_xor(v, 2, 32);
    v += __shfl_xor(v, 4, 32);
    v += __shfl_xor(v, 8, 32);
    return v;
}

// ---------------- weight cast + transpose: WT[n*K+k] = (f16)W[k*N+n] ----------------
__global__ __launch_bounds__(256) void cast_transpose(
    const float* __restrict__ W, half_t* __restrict__ WT, int K, int N) {
    size_t idx = (size_t)blockIdx.x * 256 + threadIdx.x;
    if (idx >= (size_t)K * N) return;
    int n = (int)(idx / K);
    int k = (int)(idx % K);
    WT[idx] = (half_t)W[(size_t)k * N + n];
}

// ---------------- LayerNorm (f32 in -> f16 out), one 256-thread block per row ----------------
__global__ __launch_bounds__(256) void ln_f16(
    const float* __restrict__ x, const float* __restrict__ g,
    const float* __restrict__ b, half_t* __restrict__ out) {
    const int row = blockIdx.x;
    const int tid = threadIdx.x;
    const float* xr = x + (size_t)row * D_;
    float v0 = xr[tid], v1 = xr[tid + 256], v2 = xr[tid + 512];
    __shared__ float red[256];
    red[tid] = v0 + v1 + v2;
    __syncthreads();
    for (int s = 128; s > 0; s >>= 1) { if (tid < s) red[tid] += red[tid + s]; __syncthreads(); }
    float mu = red[0] * (1.0f / D_);
    __syncthreads();
    float d0 = v0 - mu, d1 = v1 - mu, d2 = v2 - mu;
    red[tid] = d0 * d0 + d1 * d1 + d2 * d2;
    __syncthreads();
    for (int s = 128; s > 0; s >>= 1) { if (tid < s) red[tid] += red[tid + s]; __syncthreads(); }
    float inv = rsqrtf(red[0] * (1.0f / D_) + 1e-5f);
    half_t* outr = out + (size_t)row * D_;
    outr[tid]       = (half_t)(d0 * inv * g[tid]       + b[tid]);
    outr[tid + 256] = (half_t)(d1 * inv * g[tid + 256] + b[tid + 256]);
    outr[tid + 512] = (half_t)(d2 * inv * g[tid + 512] + b[tid + 512]);
}

// ---------------- V transpose: v[B,S,H,DK] -> vT[B,H,DK,S] ----------------
__global__ __launch_bounds__(256) void transpose_v(
    const half_t* __restrict__ v, half_t* __restrict__ vt) {
    size_t idx = (size_t)blockIdx.x * 256 + threadIdx.x; // over B*H*DK*S
    int s = (int)(idx % S_);
    size_t r = idx / S_;
    int d = (int)(r % DK_); r /= DK_;
    int h = (int)(r % H_);
    int b = (int)(r / H_);
    vt[idx] = v[(((size_t)b * S_ + s) * H_ + h) * DK_ + d];
}

// ---------------- WMMA GEMM: out[M,N] = A[M,K] @ Bt[N,K]^T (+epilogue) ----------------
// Block tile 128x128, 8 waves of 32x64 register tiles, LDS-staged K-steps of 32.
// EPI 0: f16 store            EPI 1: +bias, exact GELU, f16 store
// EPI 2: +resid, f32 store    EPI 3: +bias +resid, f32 store
template <int EPI>
__global__ __launch_bounds__(256) void gemm_wmma(
    const half_t* __restrict__ A, const half_t* __restrict__ Bt,
    const float* __restrict__ bias, const float* __restrict__ resid,
    void* __restrict__ outv, int N, int K) {
#ifdef USE_ASYNC_LDS
    __shared__ half_t ldsA[2][128 * 32];   // double-buffered 8KB stages
    __shared__ half_t ldsB[2][128 * 32];
#else
    __shared__ half_t ldsA[1][128 * 32];
    __shared__ half_t ldsB[1][128 * 32];
#endif
    const int tid  = threadIdx.x;
    const int w    = tid >> 5;
    const int lane = tid & 31;
    const int li = lane & 15, hi = lane >> 4;
    const int wm = w & 3, wn = w >> 2;          // wave grid 4(M) x 2(N)
    const int tm = blockIdx.x * 128;
    const int tn = blockIdx.y * 128;

    // staging map: each thread moves two 16B chunks per tile (rows r0, r0+64)
    const int r0 = tid >> 2;
    const int c0 = (tid & 3) * 8;
    const half_t* gA = A + (size_t)tm * K;
    const half_t* gB = Bt + (size_t)tn * K;

    v8f acc[2][4] = {};

    auto compute = [&](int buf) {
        const half_t* la = &ldsA[buf][(wm * 32) * 32];
        const half_t* lb = &ldsB[buf][(wn * 64) * 32];
        v16h a0 = load_frag(la, 32, 0);
        v16h a1 = load_frag(la + 16 * 32, 32, 0);
#pragma unroll
        for (int t = 0; t < 4; ++t) {
            v16h bf = load_frag(lb + t * 16 * 32, 32, 0);
            acc[0][t] = wmma_f16(a0, bf, acc[0][t]);
            acc[1][t] = wmma_f16(a1, bf, acc[1][t]);
        }
    };

#ifdef USE_ASYNC_LDS
    auto stage = [&](int buf, int kb) {
        async_cp16(gA + (size_t)r0 * K + kb + c0,        &ldsA[buf][r0 * 32 + c0]);
        async_cp16(gA + (size_t)(r0 + 64) * K + kb + c0, &ldsA[buf][(r0 + 64) * 32 + c0]);
        async_cp16(gB + (size_t)r0 * K + kb + c0,        &ldsB[buf][r0 * 32 + c0]);
        async_cp16(gB + (size_t)(r0 + 64) * K + kb + c0, &ldsB[buf][(r0 + 64) * 32 + c0]);
    };
    stage(0, 0);
    WAIT_ASYNC0();
    __syncthreads();
    int cur = 0;
    for (int kb = 0; kb < K; kb += 32) {
        const bool more = (kb + 32 < K);
        if (more) stage(cur ^ 1, kb + 32);       // prefetch next stage into other buffer
        compute(cur);
        if (more) WAIT_ASYNC0();
        __syncthreads();
        cur ^= 1;
    }
#else
    // register double-buffer: global loads for step k+1 overlap WMMAs on step k
    v8h a0r, a1r, b0r, b1r;
    auto gload = [&](int kb) {
        a0r = *(const v8h*)(gA + (size_t)r0 * K + kb + c0);
        a1r = *(const v8h*)(gA + (size_t)(r0 + 64) * K + kb + c0);
        b0r = *(const v8h*)(gB + (size_t)r0 * K + kb + c0);
        b1r = *(const v8h*)(gB + (size_t)(r0 + 64) * K + kb + c0);
    };
    gload(0);
    for (int kb = 0; kb < K; kb += 32) {
        *(v8h*)&ldsA[0][r0 * 32 + c0]        = a0r;
        *(v8h*)&ldsA[0][(r0 + 64) * 32 + c0] = a1r;
        *(v8h*)&ldsB[0][r0 * 32 + c0]        = b0r;
        *(v8h*)&ldsB[0][(r0 + 64) * 32 + c0] = b1r;
        __syncthreads();
        if (kb + 32 < K) gload(kb + 32);
        compute(0);
        __syncthreads();
    }
#endif

#pragma unroll
    for (int mi = 0; mi < 2; ++mi) {
#pragma unroll
        for (int t = 0; t < 4; ++t) {
            const int col = tn + wn * 64 + t * 16 + li;
            float bval = 0.0f;
            if constexpr (EPI == 1 || EPI == 3) bval = bias[col];
#pragma unroll
            for (int r = 0; r < 8; ++r) {
                const int row = tm + wm * 32 + mi * 16 + r + 8 * hi;
                const size_t idx = (size_t)row * N + col;
                float vv = acc[mi][t][r];
                if constexpr (EPI == 0) {
                    ((half_t*)outv)[idx] = (half_t)vv;
                } else if constexpr (EPI == 1) {
                    vv += bval;
                    vv = 0.5f * vv * (1.0f + erff(vv * 0.70710678118f)); // exact GELU
                    ((half_t*)outv)[idx] = (half_t)vv;
                } else if constexpr (EPI == 2) {
                    ((float*)outv)[idx] = vv + resid[idx];
                } else {
                    ((float*)outv)[idx] = vv + bval + resid[idx];
                }
            }
        }
    }
}

// ---------------- Flash attention: one wave per (b,h, 16-query tile) ----------------
__global__ __launch_bounds__(32) void flash_attn(
    const half_t* __restrict__ q, const half_t* __restrict__ k,
    const half_t* __restrict__ vt, half_t* __restrict__ attn) {
    __shared__ half_t pbuf[16 * 32];   // P tile bounce buffer (C-layout -> A-layout)
    const int qt = blockIdx.x;
    const int bh = blockIdx.y;
    const int b = bh / H_, h = bh % H_;
    const int lane = threadIdx.x;
    const int li = lane & 15, hi = lane >> 4;

    const half_t* qb = q + (((size_t)b * S_ + qt * 16) * H_ + h) * DK_;
    const v16h q0 = load_frag(qb, D_, 0);    // K = dk 0..31
    const v16h q1 = load_frag(qb, D_, 32);   // K = dk 32..63

    v8f o0 = {}, o1 = {}, o2 = {}, o3 = {};
    float m[8], l[8];
#pragma unroll
    for (int r = 0; r < 8; ++r) { m[r] = -1e30f; l[r] = 0.0f; }

    const half_t* vtb = vt + ((size_t)bh * DK_) * S_;  // [dk][s] for this (b,h)

    for (int j = 0; j < S_; j += 32) {
        const half_t* kb0 = k + (((size_t)b * S_ + j) * H_ + h) * DK_;
        const half_t* kb1 = kb0 + (size_t)16 * D_;
        v8f s0 = {}, s1 = {};
        s0 = wmma_f16(q0, load_frag(kb0, D_, 0),  s0);
        s0 = wmma_f16(q1, load_frag(kb0, D_, 32), s0);
        s1 = wmma_f16(q0, load_frag(kb1, D_, 0),  s1);
        s1 = wmma_f16(q1, load_frag(kb1, D_, 32), s1);

#pragma unroll
        for (int r = 0; r < 8; ++r) {
            float a0 = s0[r] * 0.125f;          // scale = DK^-0.5
            float a1 = s1[r] * 0.125f;
            float mx = hmax16(fmaxf(a0, a1));   // row max over 32 keys
            float mn = fmaxf(m[r], mx);
            float alpha = __expf(m[r] - mn);
            float p0 = __expf(a0 - mn);
            float p1 = __expf(a1 - mn);
            float rs = hsum16(p0 + p1);         // row sum over 32 keys
            l[r] = l[r] * alpha + rs;
            m[r] = mn;
            o0[r] *= alpha; o1[r] *= alpha; o2[r] *= alpha; o3[r] *= alpha;
            // store P in row-major [query 16][key 32] for A-fragment reload
            pbuf[(r + 8 * hi) * 32 + li]      = (half_t)p0;
            pbuf[(r + 8 * hi) * 32 + 16 + li] = (half_t)p1;
        }
        __syncthreads();                         // ds stores visible
        v16h pa = load_frag(pbuf, 32, 0);        // P as 16x32 A-fragment
        __syncthreads();                         // loads done before next overwrite

        o0 = wmma_f16(pa, load_frag(vtb,                   S_, j), o0);
        o1 = wmma_f16(pa, load_frag(vtb + (size_t)16 * S_, S_, j), o1);
        o2 = wmma_f16(pa, load_frag(vtb + (size_t)32 * S_, S_, j), o2);
        o3 = wmma_f16(pa, load_frag(vtb + (size_t)48 * S_, S_, j), o3);
    }

#pragma unroll
    for (int r = 0; r < 8; ++r) {
        const float inv = 1.0f / l[r];
        const int srow = qt * 16 + r + 8 * hi;
        half_t* op = attn + (((size_t)b * S_ + srow) * H_ + h) * DK_ + li;
        op[0]  = (half_t)(o0[r] * inv);
        op[16] = (half_t)(o1[r] * inv);
        op[32] = (half_t)(o2[r] * inv);
        op[48] = (half_t)(o3[r] * inv);
    }
}

// ---------------- host-side orchestration ----------------
extern "C" void kernel_launch(void* const* d_in, const int* in_sizes, int n_in,
                              void* d_out, int out_size, void* d_ws, size_t ws_size,
                              hipStream_t stream) {
    (void)in_sizes; (void)n_in; (void)out_size; (void)ws_size;
    const float* x   = (const float*)d_in[0];
    const float* Wq  = (const float*)d_in[1];
    const float* Wk  = (const float*)d_in[2];
    const float* Wv  = (const float*)d_in[3];
    const float* Wo  = (const float*)d_in[4];
    const float* g1  = (const float*)d_in[5];
    const float* b1  = (const float*)d_in[6];
    const float* g2  = (const float*)d_in[7];
    const float* b2  = (const float*)d_in[8];
    const float* W1  = (const float*)d_in[9];
    const float* bW1 = (const float*)d_in[10];
    const float* W2  = (const float*)d_in[11];
    const float* bW2 = (const float*)d_in[12];

    // workspace layout (bytes); buffers reused across phases
    constexpr size_t SZ_TD_H = NTOK * D_   * sizeof(half_t);   // 12.6 MB
    constexpr size_t SZ_TD_F = NTOK * D_   * sizeof(float);    // 25.2 MB
    constexpr size_t SZ_FF_H = NTOK * DFF_ * sizeof(half_t);   // 50.3 MB (== 4*SZ_TD_H)
    constexpr size_t SZ_W_H  = (size_t)D_ * D_   * sizeof(half_t);
    constexpr size_t SZ_W1_H = (size_t)D_ * DFF_ * sizeof(half_t);

    char* w = (char*)d_ws;
    half_t* h1   = (half_t*)(w);                                // LN1 out; reused as LN2 out
    half_t* qb   = (half_t*)(w + SZ_TD_H);                      // reused as FFN hidden
    half_t* kb   = (half_t*)(w + 2 * SZ_TD_H);
    half_t* vb   = (half_t*)(w + 3 * SZ_TD_H);
    half_t* vtb  = (half_t*)(w + 4 * SZ_TD_H);
    half_t* ffh  = qb;                                          // spans q..vT (50.3 MB)
    half_t* attn = (half_t*)(w + SZ_TD_H + SZ_FF_H);
    float*  xa   = (float*) (w + 2 * SZ_TD_H + SZ_FF_H);
    char*   wbp  = w + 2 * SZ_TD_H + SZ_FF_H + SZ_TD_F;
    half_t* WqT = (half_t*)(wbp);
    half_t* WkT = (half_t*)(wbp + SZ_W_H);
    half_t* WvT = (half_t*)(wbp + 2 * SZ_W_H);
    half_t* WoT = (half_t*)(wbp + 3 * SZ_W_H);
    half_t* W1T = (half_t*)(wbp + 4 * SZ_W_H);
    half_t* W2T = (half_t*)(wbp + 4 * SZ_W_H + SZ_W1_H);

    const dim3 blk(256);
    const int nW  = D_ * D_;
    const int nW1 = D_ * DFF_;
    cast_transpose<<<(nW  + 255) / 256, blk, 0, stream>>>(Wq, WqT, D_,  D_);
    cast_transpose<<<(nW  + 255) / 256, blk, 0, stream>>>(Wk, WkT, D_,  D_);
    cast_transpose<<<(nW  + 255) / 256, blk, 0, stream>>>(Wv, WvT, D_,  D_);
    cast_transpose<<<(nW  + 255) / 256, blk, 0, stream>>>(Wo, WoT, D_,  D_);
    cast_transpose<<<(nW1 + 255) / 256, blk, 0, stream>>>(W1, W1T, D_,  DFF_);
    cast_transpose<<<(nW1 + 255) / 256, blk, 0, stream>>>(W2, W2T, DFF_, D_);

    // pre-norm attention block
    ln_f16<<<(int)NTOK, blk, 0, stream>>>(x, g1, b1, h1);
    const dim3 g768((int)NTOK / 128, D_ / 128);     // 64 x 6
    gemm_wmma<0><<<g768, blk, 0, stream>>>(h1, WqT, nullptr, nullptr, qb, D_, D_);
    gemm_wmma<0><<<g768, blk, 0, stream>>>(h1, WkT, nullptr, nullptr, kb, D_, D_);
    gemm_wmma<0><<<g768, blk, 0, stream>>>(h1, WvT, nullptr, nullptr, vb, D_, D_);
    transpose_v<<<(int)((NTOK * D_) / 256), blk, 0, stream>>>(vb, vtb);
    flash_attn<<<dim3(S_ / 16, B_ * H_), dim3(32), 0, stream>>>(qb, kb, vtb, attn);
    gemm_wmma<2><<<g768, blk, 0, stream>>>(attn, WoT, nullptr, x, xa, D_, D_);

    // pre-norm FFN block
    ln_f16<<<(int)NTOK, blk, 0, stream>>>(xa, g2, b2, h1);
    const dim3 gff((int)NTOK / 128, DFF_ / 128);    // 64 x 24
    gemm_wmma<1><<<gff, blk, 0, stream>>>(h1, W1T, bW1, nullptr, ffh, DFF_, D_);
    gemm_wmma<3><<<g768, blk, 0, stream>>>(ffh, W2T, bW2, xa, d_out, D_, DFF_);
}
